// RAFTNet_66022237274530
// MI455X (gfx1250) — compile-verified, matrix operand order checked by reading, not run on hardware
//
#include <hip/hip_runtime.h>

// ---------------------------------------------------------------------------
// Types for CDNA5 WMMA
// ---------------------------------------------------------------------------
typedef __attribute__((ext_vector_type(16))) __bf16 v16bf;
typedef __attribute__((ext_vector_type(8)))  __bf16 v8bf;
typedef __attribute__((ext_vector_type(8)))  float  v8f;

#define TM 128          // block tile M (4 waves x 32)
#define TN 64           // block tile N (2 waves x 32)
#define LDA 40          // row pitch (elems): 80B = 5*16B -> aligned b128 frags
#define LDB 40

// Fragment per ISA 16-bit A/B layout: elems 0..7 = K[khal..khal+7],
// elems 8..15 = K[16+khal..16+khal+7]; khal = (lane<16)?0:8.
// Row base is 16B-aligned and khal*2 in {0,16} -> two aligned b128 LDS loads.
static __device__ __forceinline__ v16bf frag16(const __bf16* row, int khal)
{
    v8bf lo = *(const v8bf*)(row + khal);
    v8bf hi = *(const v8bf*)(row + 16 + khal);
    return __builtin_shufflevector(lo, hi, 0, 1, 2, 3, 4, 5, 6, 7,
                                   8, 9, 10, 11, 12, 13, 14, 15);
}

// ---------------------------------------------------------------------------
// Implicit-GEMM convolution, bf16 WMMA, f32 accumulate.
// in  : bf16 NCHW
// Bw  : bf16 [K][Cout]  (K = Cin*KH*KW), repacked weights
// out : f32  NCHW (+bias, *alpha, optional relu)
// 256 threads = 8 waves (4x2); block tile 128(M) x 64(N); wave tile 32x32
// = 2x2 WMMA tiles -> 4 v_wmma per K-step with 2x register reuse of frags.
// ---------------------------------------------------------------------------
__global__ void k_conv_wmma(const __bf16* __restrict__ in,
                            const __bf16* __restrict__ Bw,
                            const float* __restrict__ bias,
                            float* __restrict__ out,
                            int Nb, int Cin, int Hin, int Win,
                            int Cout, int Hout, int Wout,
                            int KH, int KW, int stride, int padH, int padW,
                            int relu, float alpha)
{
    const int Ktot = Cin * KH * KW;
    const int M    = Nb * Hout * Wout;
    const int mBlk = blockIdx.x * TM;
    const int nBlk = blockIdx.y * TN;

    __shared__ __align__(16) __bf16 As[TM][LDA];   // [m][k]
    __shared__ __align__(16) __bf16 Bt[TN][LDB];   // [n][k] (transposed)

    const int tid  = threadIdx.x;
    const int wave = tid >> 5;
    const int lane = tid & 31;
    const int wm   = (wave & 3) * 32;
    const int wn   = (wave >> 2) * 32;
    const int l15  = lane & 15;
    const int khal = (lane < 16) ? 0 : 8;

    v8f acc[2][2];
#pragma unroll
    for (int i = 0; i < 2; ++i)
#pragma unroll
        for (int j = 0; j < 2; ++j)
            acc[i][j] = (v8f){0.f,0.f,0.f,0.f,0.f,0.f,0.f,0.f};

    for (int k0 = 0; k0 < Ktot; k0 += 32) {
        // ---- stage A tile (im2col on the fly): TM x 32 ----
        for (int e = tid; e < TM * 32; e += 256) {
            int m  = e >> 5;
            int kk = e & 31;
            int mg = mBlk + m;
            int kg = k0 + kk;
            __bf16 val = (__bf16)0.f;
            if (mg < M && kg < Ktot) {
                int ow = mg % Wout; int t = mg / Wout;
                int oh = t % Hout;  int nb = t / Hout;
                int kw = kg % KW;   int t2 = kg / KW;
                int kh = t2 % KH;   int ci = t2 / KH;
                int ih = oh * stride - padH + kh;
                int iw = ow * stride - padW + kw;
                if (ih >= 0 && ih < Hin && iw >= 0 && iw < Win)
                    val = in[(((long)nb * Cin + ci) * Hin + ih) * Win + iw];
            }
            As[m][kk] = val;
        }
        // ---- stage B tile transposed: Bt[n][k], global reads coalesced in n ----
        for (int e = tid; e < 32 * TN; e += 256) {
            int kk = e / TN;
            int n  = e % TN;
            int kg = k0 + kk;
            int ng = nBlk + n;
            Bt[n][kk] = (kg < Ktot && ng < Cout)
                          ? Bw[(long)kg * Cout + ng] : (__bf16)0.f;
        }
        if (lane == 0 && (k0 + 32) < Ktot)
            __builtin_prefetch(Bw + (long)(k0 + 32) * Cout + nBlk, 0, 1);
        __syncthreads();

        v16bf a0 = frag16(As[wm + l15],      khal);
        v16bf a1 = frag16(As[wm + 16 + l15], khal);
        v16bf b0 = frag16(Bt[wn + l15],      khal);
        v16bf b1 = frag16(Bt[wn + 16 + l15], khal);

        acc[0][0] = __builtin_amdgcn_wmma_f32_16x16x32_bf16(
                        false, a0, false, b0, (short)0, acc[0][0], false, false);
        acc[0][1] = __builtin_amdgcn_wmma_f32_16x16x32_bf16(
                        false, a0, false, b1, (short)0, acc[0][1], false, false);
        acc[1][0] = __builtin_amdgcn_wmma_f32_16x16x32_bf16(
                        false, a1, false, b0, (short)0, acc[1][0], false, false);
        acc[1][1] = __builtin_amdgcn_wmma_f32_16x16x32_bf16(
                        false, a1, false, b1, (short)0, acc[1][1], false, false);
        __syncthreads();
    }

    // ---- epilogue: C/D layout (VGPR r: lanes0-15 M=r, lanes16-31 M=r+8) ----
#pragma unroll
    for (int sm = 0; sm < 2; ++sm)
#pragma unroll
        for (int sn = 0; sn < 2; ++sn) {
            int ng = nBlk + wn + sn * 16 + l15;
#pragma unroll
            for (int r = 0; r < 8; ++r) {
                int mg = mBlk + wm + sm * 16 + r + ((lane < 16) ? 0 : 8);
                if (mg < M && ng < Cout) {
                    float v = (acc[sm][sn][r] + bias[ng]) * alpha;
                    if (relu) v = fmaxf(v, 0.f);
                    int ow = mg % Wout; int t = mg / Wout;
                    int oh = t % Hout;  int nb = t / Hout;
                    out[(((long)nb * Cout + ng) * Hout + oh) * Wout + ow] = v;
                }
            }
        }
}

// ---------------------------------------------------------------------------
// C[b][m][n] = alpha * sum_k A[b][k][m] * B[b][k][n]   (correlation volume)
// A,B bf16 row-major (K x M / K x N) per batch; grid.z = batch.
// Same 128x64 block / 32x32 wave register tiling as the conv kernel.
// ---------------------------------------------------------------------------
__global__ void k_gemm_atb_wmma(const __bf16* __restrict__ Ag,
                                const __bf16* __restrict__ Bg,
                                float* __restrict__ Cg,
                                int M, int N, int K, float alpha)
{
    const __bf16* A = Ag + (long)blockIdx.z * K * M;
    const __bf16* B = Bg + (long)blockIdx.z * K * N;
    float*        C = Cg + (long)blockIdx.z * M * N;

    const int mBlk = blockIdx.x * TM;
    const int nBlk = blockIdx.y * TN;

    __shared__ __align__(16) __bf16 As[TM][LDA];
    __shared__ __align__(16) __bf16 Bt[TN][LDB];

    const int tid  = threadIdx.x;
    const int wave = tid >> 5;
    const int lane = tid & 31;
    const int wm   = (wave & 3) * 32;
    const int wn   = (wave >> 2) * 32;
    const int l15  = lane & 15;
    const int khal = (lane < 16) ? 0 : 8;

    v8f acc[2][2];
#pragma unroll
    for (int i = 0; i < 2; ++i)
#pragma unroll
        for (int j = 0; j < 2; ++j)
            acc[i][j] = (v8f){0.f,0.f,0.f,0.f,0.f,0.f,0.f,0.f};

    for (int k0 = 0; k0 < K; k0 += 32) {
        // A staged m-minor: consecutive lanes read consecutive m (coalesced)
        for (int e = tid; e < TM * 32; e += 256) {
            int kk = e >> 7;            // TM = 128
            int m  = e & 127;
            int mg = mBlk + m, kg = k0 + kk;
            As[m][kk] = (mg < M && kg < K) ? A[(long)kg * M + mg] : (__bf16)0.f;
        }
        for (int e = tid; e < 32 * TN; e += 256) {
            int kk = e / TN, n = e % TN;
            int kg = k0 + kk, ng = nBlk + n;
            Bt[n][kk] = (kg < K && ng < N) ? B[(long)kg * N + ng] : (__bf16)0.f;
        }
        if (lane == 0 && (k0 + 32) < K)
            __builtin_prefetch(B + (long)(k0 + 32) * N + nBlk, 0, 1);
        __syncthreads();

        v16bf a0 = frag16(As[wm + l15],      khal);
        v16bf a1 = frag16(As[wm + 16 + l15], khal);
        v16bf b0 = frag16(Bt[wn + l15],      khal);
        v16bf b1 = frag16(Bt[wn + 16 + l15], khal);

        acc[0][0] = __builtin_amdgcn_wmma_f32_16x16x32_bf16(
                        false, a0, false, b0, (short)0, acc[0][0], false, false);
        acc[0][1] = __builtin_amdgcn_wmma_f32_16x16x32_bf16(
                        false, a0, false, b1, (short)0, acc[0][1], false, false);
        acc[1][0] = __builtin_amdgcn_wmma_f32_16x16x32_bf16(
                        false, a1, false, b0, (short)0, acc[1][0], false, false);
        acc[1][1] = __builtin_amdgcn_wmma_f32_16x16x32_bf16(
                        false, a1, false, b1, (short)0, acc[1][1], false, false);
        __syncthreads();
    }

#pragma unroll
    for (int sm = 0; sm < 2; ++sm)
#pragma unroll
        for (int sn = 0; sn < 2; ++sn) {
            int ng = nBlk + wn + sn * 16 + l15;
#pragma unroll
            for (int r = 0; r < 8; ++r) {
                int mg = mBlk + wm + sm * 16 + r + ((lane < 16) ? 0 : 8);
                if (mg < M && ng < N)
                    C[(long)mg * N + ng] = acc[sm][sn][r] * alpha;
            }
        }
}

// ---------------------------------------------------------------------------
// Small utility kernels
// ---------------------------------------------------------------------------
__global__ void k_f32_to_bf16(const float* __restrict__ x, __bf16* __restrict__ o, long n)
{
    long i = (long)blockIdx.x * blockDim.x + threadIdx.x;
    if (i < n) o[i] = (__bf16)x[i];
}

__global__ void k_img_to_bf16(const float* __restrict__ x, __bf16* __restrict__ o, long n)
{
    long i = (long)blockIdx.x * blockDim.x + threadIdx.x;
    if (i < n) o[i] = (__bf16)(2.f * x[i] - 1.f);
}

// OIHW (Cout x K) -> [K][Cout] bf16
__global__ void k_repack(const float* __restrict__ w, __bf16* __restrict__ o, int Cout, int K)
{
    long i = (long)blockIdx.x * blockDim.x + threadIdx.x;
    if (i < (long)K * Cout) {
        int k = (int)(i / Cout), c = (int)(i % Cout);
        o[i] = (__bf16)w[(long)c * K + k];
    }
}

__global__ void k_instance_norm(float* __restrict__ x, long HW, int relu)
{
    long base = (long)blockIdx.x * HW;
    __shared__ float rs[256], rq[256];
    float s = 0.f, q = 0.f;
    for (long i = threadIdx.x; i < HW; i += 256) { float v = x[base + i]; s += v; q += v * v; }
    rs[threadIdx.x] = s; rq[threadIdx.x] = q; __syncthreads();
    for (int o = 128; o > 0; o >>= 1) {
        if ((int)threadIdx.x < o) { rs[threadIdx.x] += rs[threadIdx.x + o]; rq[threadIdx.x] += rq[threadIdx.x + o]; }
        __syncthreads();
    }
    float mean = rs[0] / (float)HW;
    float var  = rq[0] / (float)HW - mean * mean;
    float inv  = rsqrtf(var + 1e-5f);
    for (long i = threadIdx.x; i < HW; i += 256) {
        float v = (x[base + i] - mean) * inv;
        if (relu) v = fmaxf(v, 0.f);
        x[base + i] = v;
    }
}

__global__ void k_batch_norm(float* __restrict__ x, const float* __restrict__ g,
                             const float* __restrict__ b, int N, int C, long HW, int relu)
{
    int c = blockIdx.x;
    __shared__ float rs[256], rq[256];
    float s = 0.f, q = 0.f;
    for (int n = 0; n < N; ++n) {
        const float* p = x + ((long)n * C + c) * HW;
        for (long i = threadIdx.x; i < HW; i += 256) { float v = p[i]; s += v; q += v * v; }
    }
    rs[threadIdx.x] = s; rq[threadIdx.x] = q; __syncthreads();
    for (int o = 128; o > 0; o >>= 1) {
        if ((int)threadIdx.x < o) { rs[threadIdx.x] += rs[threadIdx.x + o]; rq[threadIdx.x] += rq[threadIdx.x + o]; }
        __syncthreads();
    }
    float cnt  = (float)N * (float)HW;
    float mean = rs[0] / cnt;
    float var  = rq[0] / cnt - mean * mean;
    float inv  = rsqrtf(var + 1e-5f);
    float gg = g[c], bb = b[c];
    for (int n = 0; n < N; ++n) {
        float* p = x + ((long)n * C + c) * HW;
        for (long i = threadIdx.x; i < HW; i += 256) {
            float v = (p[i] - mean) * inv * gg + bb;
            if (relu) v = fmaxf(v, 0.f);
            p[i] = v;
        }
    }
}

__global__ void k_add_relu(float* __restrict__ a, const float* __restrict__ b, long n)
{
    long i = (long)blockIdx.x * blockDim.x + threadIdx.x;
    if (i < n) a[i] = fmaxf(a[i] + b[i], 0.f);
}

__global__ void k_add(float* __restrict__ a, const float* __restrict__ b, long n)
{
    long i = (long)blockIdx.x * blockDim.x + threadIdx.x;
    if (i < n) a[i] += b[i];
}

__global__ void k_sigmoid(float* __restrict__ x, long n)
{
    long i = (long)blockIdx.x * blockDim.x + threadIdx.x;
    if (i < n) x[i] = 1.f / (1.f + __expf(-x[i]));
}

__global__ void k_tanh(float* __restrict__ x, long n)
{
    long i = (long)blockIdx.x * blockDim.x + threadIdx.x;
    if (i < n) x[i] = tanhf(x[i]);
}

__global__ void k_mul(const float* __restrict__ a, const float* __restrict__ b,
                      float* __restrict__ o, long n)
{
    long i = (long)blockIdx.x * blockDim.x + threadIdx.x;
    if (i < n) o[i] = a[i] * b[i];
}

// h = (1-z)*h + z*q
__global__ void k_gru_combine(float* __restrict__ h, const float* __restrict__ z,
                              const float* __restrict__ q, long n)
{
    long i = (long)blockIdx.x * blockDim.x + threadIdx.x;
    if (i < n) h[i] = (1.f - z[i]) * h[i] + z[i] * q[i];
}

__global__ void k_concat2_bf16(const float* __restrict__ a, const float* __restrict__ b,
                               __bf16* __restrict__ o, int B, int Ca, int Cb, long HW)
{
    long n = (long)B * (Ca + Cb) * HW;
    long i = (long)blockIdx.x * blockDim.x + threadIdx.x;
    if (i >= n) return;
    long ii = i % HW; long t = i / HW;
    int c = (int)(t % (Ca + Cb)); int bb = (int)(t / (Ca + Cb));
    float v = (c < Ca) ? a[((long)bb * Ca + c) * HW + ii]
                       : b[((long)bb * Cb + (c - Ca)) * HW + ii];
    o[i] = (__bf16)v;
}

__global__ void k_concat3_f32(const float* __restrict__ a, const float* __restrict__ b,
                              const float* __restrict__ c, float* __restrict__ o,
                              int B, int Ca, int Cb, int Cc, long HW)
{
    int Ct = Ca + Cb + Cc;
    long n = (long)B * Ct * HW;
    long i = (long)blockIdx.x * blockDim.x + threadIdx.x;
    if (i >= n) return;
    long ii = i % HW; long t = i / HW;
    int ch = (int)(t % Ct); int bb = (int)(t / Ct);
    float v;
    if (ch < Ca)            v = a[((long)bb * Ca + ch) * HW + ii];
    else if (ch < Ca + Cb)  v = b[((long)bb * Cb + (ch - Ca)) * HW + ii];
    else                    v = c[((long)bb * Cc + (ch - Ca - Cb)) * HW + ii];
    o[i] = v;
}

__global__ void k_split_tanh_relu(const float* __restrict__ cn, float* __restrict__ net,
                                  float* __restrict__ inp, int B, long HW)
{
    long n = (long)B * 128 * HW;
    long i = (long)blockIdx.x * blockDim.x + threadIdx.x;
    if (i >= n) return;
    long ii = i % HW; long t = i / HW;
    int c = (int)(t % 128); int b = (int)(t / 128);
    net[i] = tanhf(cn[((long)b * 256 + c) * HW + ii]);
    inp[i] = fmaxf(cn[((long)b * 256 + 128 + c) * HW + ii], 0.f);
}

__global__ void k_init_coords(float* __restrict__ c, int B, int H, int W)
{
    long n = (long)B * 2 * H * W;
    long i = (long)blockIdx.x * blockDim.x + threadIdx.x;
    if (i >= n) return;
    int w = (int)(i % W); long t = i / W;
    int h = (int)(t % H); t /= H;
    int ch = (int)(t % 2);
    c[i] = (ch == 0) ? (float)w : (float)h;
}

__global__ void k_flow_from_coords(const float* __restrict__ c, float* __restrict__ f,
                                   int B, int H, int W)
{
    long n = (long)B * 2 * H * W;
    long i = (long)blockIdx.x * blockDim.x + threadIdx.x;
    if (i >= n) return;
    int w = (int)(i % W); long t = i / W;
    int h = (int)(t % H); t /= H;
    int ch = (int)(t % 2);
    f[i] = c[i] - ((ch == 0) ? (float)w : (float)h);
}

__global__ void k_avgpool2(const float* __restrict__ in, float* __restrict__ out,
                           long nimg, int h, int w)
{
    int ho = h >> 1, wo = w >> 1;
    long n = nimg * ho * wo;
    long i = (long)blockIdx.x * blockDim.x + threadIdx.x;
    if (i >= n) return;
    int x = (int)(i % wo); long t = i / wo;
    int y = (int)(t % ho); long im = t / ho;
    const float* p = in + ((long)im * h + 2 * y) * w + 2 * x;
    out[i] = 0.25f * (p[0] + p[1] + p[w] + p[w + 1]);
}

__device__ __forceinline__ float bilin(const float* img, int Hl, int Wl, float x, float y)
{
    float x0 = floorf(x), y0 = floorf(y);
    float wx = x - x0, wy = y - y0;
    float acc = 0.f;
#pragma unroll
    for (int dy = 0; dy < 2; ++dy)
#pragma unroll
        for (int dx = 0; dx < 2; ++dx) {
            float xi = x0 + dx, yi = y0 + dy;
            bool valid = (xi >= 0.f) && (xi <= (float)(Wl - 1)) &&
                         (yi >= 0.f) && (yi <= (float)(Hl - 1));
            int xc = min(max((int)fminf(fmaxf(xi, 0.f), (float)(Wl - 1)), 0), Wl - 1);
            int yc = min(max((int)fminf(fmaxf(yi, 0.f), (float)(Hl - 1)), 0), Hl - 1);
            float g = img[(long)yc * Wl + xc];
            float wgt = (dx ? wx : 1.f - wx) * (dy ? wy : 1.f - wy);
            acc += valid ? g * wgt : 0.f;
        }
    return acc;
}

__global__ void k_corr_lookup(const float* __restrict__ p0, const float* __restrict__ p1,
                              const float* __restrict__ p2, const float* __restrict__ p3,
                              const float* __restrict__ coords, float* __restrict__ out,
                              int B, int H, int W)
{
    long n = (long)B * 324 * H * W;
    long i = (long)blockIdx.x * blockDim.x + threadIdx.x;
    if (i >= n) return;
    int w = (int)(i % W); long t = i / W;
    int h = (int)(t % H); t /= H;
    int ch = (int)(t % 324); int b = (int)(t / 324);
    int l = ch / 81; int r2 = ch % 81;
    int dyi = r2 / 9, dxi = r2 % 9;
    float cx = coords[(((long)b * 2 + 0) * H + h) * W + w];
    float cy = coords[(((long)b * 2 + 1) * H + h) * W + w];
    float sc = 1.f / (float)(1 << l);
    float x = cx * sc + (float)(dxi - 4);
    float y = cy * sc + (float)(dyi - 4);
    int Hl = H >> l, Wl = W >> l;
    const float* img = (l == 0) ? p0 : (l == 1) ? p1 : (l == 2) ? p2 : p3;
    img += (long)(((long)b * H + h) * W + w) * Hl * Wl;
    out[i] = bilin(img, Hl, Wl, x, y);
}

__global__ void k_upsample(const float* __restrict__ flow, const float* __restrict__ mask,
                           float* __restrict__ out, int B, int H, int W)
{
    long n = (long)B * 2 * 8 * H * 8 * W;
    long i = (long)blockIdx.x * blockDim.x + threadIdx.x;
    if (i >= n) return;
    int x = (int)(i % (8 * W)); long t = i / (8 * W);
    int y = (int)(t % (8 * H)); t /= (8 * H);
    int c = (int)(t % 2); int b = (int)(t / 2);
    int ow = x >> 3, xx = x & 7, oh = y >> 3, yy = y & 7;
    float m[9]; float mx = -1e30f;
#pragma unroll
    for (int k = 0; k < 9; ++k) {
        m[k] = mask[(((long)b * 576 + k * 64 + yy * 8 + xx) * H + oh) * W + ow];
        mx = fmaxf(mx, m[k]);
    }
    float den = 0.f;
#pragma unroll
    for (int k = 0; k < 9; ++k) { m[k] = __expf(m[k] - mx); den += m[k]; }
    float acc = 0.f;
#pragma unroll
    for (int k = 0; k < 9; ++k) {
        int ii = k / 3, jj = k % 3;
        int hh = oh + ii - 1, ww = ow + jj - 1;
        float v = (hh >= 0 && hh < H && ww >= 0 && ww < W)
                    ? 8.f * flow[(((long)b * 2 + c) * H + hh) * W + ww] : 0.f;
        acc += (m[k] / den) * v;
    }
    out[i] = acc;
}

// ---------------------------------------------------------------------------
// Host-side orchestration
// ---------------------------------------------------------------------------
struct Arena { char* base; size_t off; size_t cap; };
static void* amal(Arena& a, size_t bytes) {
    a.off = (a.off + 255) & ~(size_t)255;
    void* p = a.base + a.off;
    a.off += bytes;
    return p;
}
static float*  aF(Arena& a, size_t n) { return (float*)amal(a, n * 4); }
static __bf16* aB(Arena& a, size_t n) { return (__bf16*)amal(a, n * 2); }
static inline int cdiv(long n, int b) { return (int)((n + b - 1) / b); }

#define ELW(n) dim3((unsigned)cdiv((long)(n), 256)), dim3(256), 0, stream

static void to_bf16(hipStream_t stream, const float* x, __bf16* o, long n)
{ k_f32_to_bf16<<<ELW(n)>>>(x, o, n); }

static void conv(hipStream_t stream, const __bf16* in, const __bf16* W, const float* bias,
                 float* out, int Nb, int Cin, int Hin, int Win, int Cout,
                 int KH, int KW, int st, int padH, int padW, int relu, float alpha)
{
    int Hout = (Hin + 2 * padH - KH) / st + 1;
    int Wout = (Win + 2 * padW - KW) / st + 1;
    long M = (long)Nb * Hout * Wout;
    dim3 g((unsigned)cdiv(M, TM), (unsigned)cdiv(Cout, TN), 1);
    k_conv_wmma<<<g, dim3(256), 0, stream>>>(in, W, bias, out, Nb, Cin, Hin, Win,
                                             Cout, Hout, Wout, KH, KW, st, padH, padW,
                                             relu, alpha);
}

static void run_norm(hipStream_t stream, float* x, bool bn, const float* g, const float* b,
                     int N, int C, long HW, int relu)
{
    if (bn) k_batch_norm<<<dim3(C), dim3(256), 0, stream>>>(x, g, b, N, C, HW, relu);
    else    k_instance_norm<<<dim3(N * C), dim3(256), 0, stream>>>(x, HW, relu);
}

// ------------------- parameter pytree parsing (alphabetical) ----------------
struct BlkP {
    const float *c1_w, *c1_b, *c2_w, *c2_b, *dw_w, *dw_b;
    const float *n1_g, *n1_b, *n2_g, *n2_b, *n3_g, *n3_b;
    bool dw;
};
struct EncP { const float *c1_w, *c1_b, *c2_w, *c2_b, *n1_g, *n1_b; BlkP b[6]; };
struct UpdP {
    const float *cc1_w,*cc1_b,*cc2_w,*cc2_b,*cf1_w,*cf1_b,*cf2_w,*cf2_b,*cm_w,*cm_b;
    const float *fh1_w,*fh1_b,*fh2_w,*fh2_b,*mk1_w,*mk1_b,*mk2_w,*mk2_b;
    const float *qh_w,*qh_b,*qv_w,*qv_b,*rh_w,*rh_b,*rv_w,*rv_b,*zh_w,*zh_b,*zv_w,*zv_b;
};

static const int BCI[6] = {64, 64, 64, 96, 96, 128};
static const int BCO[6] = {64, 64, 96, 96, 128, 128};
static const int BST[6] = {1, 1, 2, 1, 2, 1};
static const int BHI[6] = {192, 192, 192, 96, 96, 48};
static const int BWI[6] = {256, 256, 256, 128, 128, 64};
static const bool BDW[6] = {false, false, true, false, true, false};

static int parse_enc(void* const* din, int idx, EncP& E, bool bn)
{
    auto F = [&](int i) { return (const float*)din[i]; };
    E.c1_b = F(idx++); E.c1_w = F(idx++); E.c2_b = F(idx++); E.c2_w = F(idx++);
    for (int i = 0; i < 6; ++i) {
        BlkP& B = E.b[i]; B.dw = BDW[i];
        B.c1_b = F(idx++); B.c1_w = F(idx++); B.c2_b = F(idx++); B.c2_w = F(idx++);
        if (B.dw) { B.dw_b = F(idx++); B.dw_w = F(idx++); }
        if (bn) {
            B.n1_b = F(idx++); B.n1_g = F(idx++);
            B.n2_b = F(idx++); B.n2_g = F(idx++);
            if (B.dw) { B.n3_b = F(idx++); B.n3_g = F(idx++); }
        }
    }
    if (bn) { E.n1_b = F(idx++); E.n1_g = F(idx++); }
    return idx;
}

static int parse_upd(void* const* din, int idx, UpdP& U)
{
    auto F = [&](int i) { return (const float*)din[i]; };
    U.cc1_b=F(idx++); U.cc1_w=F(idx++); U.cc2_b=F(idx++); U.cc2_w=F(idx++);
    U.cf1_b=F(idx++); U.cf1_w=F(idx++); U.cf2_b=F(idx++); U.cf2_w=F(idx++);
    U.cm_b =F(idx++); U.cm_w =F(idx++);
    U.fh1_b=F(idx++); U.fh1_w=F(idx++); U.fh2_b=F(idx++); U.fh2_w=F(idx++);
    U.mk1_b=F(idx++); U.mk1_w=F(idx++); U.mk2_b=F(idx++); U.mk2_w=F(idx++);
    U.qh_b =F(idx++); U.qh_w =F(idx++); U.qv_b =F(idx++); U.qv_w =F(idx++);
    U.rh_b =F(idx++); U.rh_w =F(idx++); U.rv_b =F(idx++); U.rv_w =F(idx++);
    U.zh_b =F(idx++); U.zh_w =F(idx++); U.zv_b =F(idx++); U.zv_w =F(idx++);
    return idx;
}

static const __bf16* prep(hipStream_t stream, Arena& ar, const float* w, int Cout, int K)
{
    __bf16* o = aB(ar, (size_t)K * Cout);
    long n = (long)K * Cout;
    k_repack<<<ELW(n)>>>(w, o, Cout, K);
    return o;
}

struct EncW { const __bf16 *c1, *c2, *b_c1[6], *b_c2[6], *b_dw[6]; };
struct UpdW {
    const __bf16 *cc1,*cc2,*cf1,*cf2,*cm,*fh1,*fh2,*mk1,*mk2;
    const __bf16 *zh,*rh,*qh,*zv,*rv,*qv;
};

static EncW prep_enc(hipStream_t stream, Arena& ar, const EncP& E)
{
    EncW W;
    W.c1 = prep(stream, ar, E.c1_w, 64, 3 * 49);
    for (int i = 0; i < 6; ++i) {
        W.b_c1[i] = prep(stream, ar, E.b[i].c1_w, BCO[i], BCI[i] * 9);
        W.b_c2[i] = prep(stream, ar, E.b[i].c2_w, BCO[i], BCO[i] * 9);
        W.b_dw[i] = BDW[i] ? prep(stream, ar, E.b[i].dw_w, BCO[i], BCI[i]) : nullptr;
    }
    W.c2 = prep(stream, ar, E.c2_w, 256, 128);
    return W;
}

static UpdW prep_upd(hipStream_t stream, Arena& ar, const UpdP& U)
{
    UpdW W;
    W.cc1 = prep(stream, ar, U.cc1_w, 256, 324);
    W.cc2 = prep(stream, ar, U.cc2_w, 192, 256 * 9);
    W.cf1 = prep(stream, ar, U.cf1_w, 128, 2 * 49);
    W.cf2 = prep(stream, ar, U.cf2_w, 64, 128 * 9);
    W.cm  = prep(stream, ar, U.cm_w, 126, 256 * 9);
    W.fh1 = prep(stream, ar, U.fh1_w, 256, 128 * 9);
    W.fh2 = prep(stream, ar, U.fh2_w, 2, 256 * 9);
    W.mk1 = prep(stream, ar, U.mk1_w, 256, 128 * 9);
    W.mk2 = prep(stream, ar, U.mk2_w, 576, 256);
    W.zh = prep(stream, ar, U.zh_w, 128, 384 * 5);
    W.rh = prep(stream, ar, U.rh_w, 128, 384 * 5);
    W.qh = prep(stream, ar, U.qh_w, 128, 384 * 5);
    W.zv = prep(stream, ar, U.zv_w, 128, 384 * 5);
    W.rv = prep(stream, ar, U.rv_w, 128, 384 * 5);
    W.qv = prep(stream, ar, U.qv_w, 128, 384 * 5);
    return W;
}

// ------------------------------- encoder ------------------------------------
static void run_encoder(hipStream_t stream, Arena ar /*scratch: by value*/,
                        const float* img, const EncP& P, const EncW& W, bool bn,
                        float* out /* (2,256,48,64) */)
{
    const int Nb = 2;
    const size_t SMAX = (size_t)2 * 64 * 192 * 256; // largest activation
    __bf16* XB = aB(ar, SMAX);
    __bf16* YB = aB(ar, SMAX);
    float*  X  = aF(ar, SMAX);
    float*  Y1 = aF(ar, SMAX);
    float*  Y2 = aF(ar, SMAX);
    float*  XS = aF(ar, SMAX);

    // stem: conv 3->64, 7x7 s2 p3 + norm + relu
    k_img_to_bf16<<<ELW((long)Nb * 3 * 384 * 512)>>>(img, XB, (long)Nb * 3 * 384 * 512);
    conv(stream, XB, W.c1, P.c1_b, X, Nb, 3, 384, 512, 64, 7, 7, 2, 3, 3, 0, 1.f);
    run_norm(stream, X, bn, P.n1_g, P.n1_b, Nb, 64, (long)192 * 256, 1);

    for (int i = 0; i < 6; ++i) {
        int ci = BCI[i], co = BCO[i], st = BST[i];
        int Hin = BHI[i], Win = BWI[i];
        int Ho = Hin / st, Wo = Win / st;
        long nin  = (long)Nb * ci * Hin * Win;
        long nout = (long)Nb * co * Ho * Wo;
        const BlkP& B = P.b[i];

        to_bf16(stream, X, XB, nin);
        conv(stream, XB, W.b_c1[i], B.c1_b, Y1, Nb, ci, Hin, Win, co, 3, 3, st, 1, 1, 0, 1.f);
        run_norm(stream, Y1, bn, B.n1_g, B.n1_b, Nb, co, (long)Ho * Wo, 1);
        to_bf16(stream, Y1, YB, nout);
        conv(stream, YB, W.b_c2[i], B.c2_b, Y2, Nb, co, Ho, Wo, co, 3, 3, 1, 1, 1, 0, 1.f);
        run_norm(stream, Y2, bn, B.n2_g, B.n2_b, Nb, co, (long)Ho * Wo, 1);
        if (B.dw) {
            conv(stream, XB, W.b_dw[i], B.dw_b, XS, Nb, ci, Hin, Win, co, 1, 1, st, 0, 0, 0, 1.f);
            run_norm(stream, XS, bn, B.n3_g, B.n3_b, Nb, co, (long)Ho * Wo, 0);
            k_add_relu<<<ELW(nout)>>>(Y2, XS, nout);
        } else {
            k_add_relu<<<ELW(nout)>>>(Y2, X, nout);
        }
        float* tmp = X; X = Y2; Y2 = tmp;
    }

    // head: 1x1 conv 128 -> 256
    to_bf16(stream, X, XB, (long)Nb * 128 * 48 * 64);
    conv(stream, XB, W.c2, P.c2_b, out, Nb, 128, 48, 64, 256, 1, 1, 1, 0, 0, 0, 1.f);
}

// ---------------------------------------------------------------------------
extern "C" void kernel_launch(void* const* d_in, const int* in_sizes, int n_in,
                              void* d_out, int out_size, void* d_ws, size_t ws_size,
                              hipStream_t stream)
{
    (void)in_sizes; (void)n_in; (void)out_size;
    const float* img1 = (const float*)d_in[0];
    const float* img2 = (const float*)d_in[1];

    EncP cnetP, fnetP; UpdP U;
    int idx = 2;
    idx = parse_enc(d_in, idx, cnetP, true);   // 'cnet' < 'fnet' < 'update'
    idx = parse_enc(d_in, idx, fnetP, false);
    idx = parse_upd(d_in, idx, U);
    const int iters = 12; // scalar input unreadable under graph capture; fixed by setup

    Arena ar{(char*)d_ws, 0, ws_size};

    // --- prepacked bf16 weights (persistent) ---
    EncW cW = prep_enc(stream, ar, cnetP);
    EncW fW = prep_enc(stream, ar, fnetP);
    UpdW uW = prep_upd(stream, ar, U);

    // --- persistent activations ---
    const int Bn = 2, H = 48, Wd = 64;
    const long HW = (long)H * Wd;           // 3072
    float* f1  = aF(ar, (size_t)Bn * 256 * HW);
    float* f2  = aF(ar, (size_t)Bn * 256 * HW);
    float* cn  = aF(ar, (size_t)Bn * 256 * HW);
    __bf16* f1b = aB(ar, (size_t)Bn * 256 * HW);
    __bf16* f2b = aB(ar, (size_t)Bn * 256 * HW);
    float* corr = aF(ar, (size_t)Bn * HW * HW);             // level 0
    float* p1   = aF(ar, (size_t)Bn * HW * (HW / 4));
    float* p2   = aF(ar, (size_t)Bn * HW * (HW / 16));
    float* p3   = aF(ar, (size_t)Bn * HW * (HW / 64));
    float* net  = aF(ar, (size_t)Bn * 128 * HW);
    float* inp  = aF(ar, (size_t)Bn * 128 * HW);
    float* coords = aF(ar, (size_t)Bn * 2 * HW);
    float* cfeat  = aF(ar, (size_t)Bn * 324 * HW);
    __bf16* cfeatb = aB(ar, (size_t)Bn * 324 * HW);
    float* c256 = aF(ar, (size_t)Bn * 256 * HW);
    __bf16* tb256 = aB(ar, (size_t)Bn * 256 * HW);
    float* c192 = aF(ar, (size_t)Bn * 192 * HW);
    float* flow = aF(ar, (size_t)Bn * 2 * HW);
    __bf16* flowb = aB(ar, (size_t)Bn * 2 * HW);
    float* fbuf128 = aF(ar, (size_t)Bn * 128 * HW);
    __bf16* tb128 = aB(ar, (size_t)Bn * 128 * HW);
    float* fbuf64 = aF(ar, (size_t)Bn * 64 * HW);
    __bf16* cat256b = aB(ar, (size_t)Bn * 256 * HW);
    float* m126 = aF(ar, (size_t)Bn * 126 * HW);
    float* xf   = aF(ar, (size_t)Bn * 256 * HW);
    __bf16* hxb = aB(ar, (size_t)Bn * 384 * HW);
    float* zb = aF(ar, (size_t)Bn * 128 * HW);
    float* rb = aF(ar, (size_t)Bn * 128 * HW);
    float* qb = aF(ar, (size_t)Bn * 128 * HW);
    float* rhb = aF(ar, (size_t)Bn * 128 * HW);
    float* t256 = aF(ar, (size_t)Bn * 256 * HW);
    float* delta = aF(ar, (size_t)Bn * 2 * HW);
    float* mask  = aF(ar, (size_t)Bn * 576 * HW);
    __bf16* netb = aB(ar, (size_t)Bn * 128 * HW);

    // --- encoders (scratch region reused per call; Arena passed by value) ---
    run_encoder(stream, ar, img1, fnetP, fW, false, f1);
    run_encoder(stream, ar, img2, fnetP, fW, false, f2);
    run_encoder(stream, ar, img1, cnetP, cW, true,  cn);

    // --- all-pairs correlation: corr[b, hw1, hw2] = f1^T f2 / sqrt(256) ---
    to_bf16(stream, f1, f1b, (long)Bn * 256 * HW);
    to_bf16(stream, f2, f2b, (long)Bn * 256 * HW);
    {
        dim3 g((unsigned)cdiv(HW, TM), (unsigned)cdiv(HW, TN), Bn);
        k_gemm_atb_wmma<<<g, dim3(256), 0, stream>>>(f1b, f2b, corr,
                                                     (int)HW, (int)HW, 256, 0.0625f);
    }
    // pyramid (avg-pool last two dims of (B*HW, H, W))
    k_avgpool2<<<ELW((long)Bn * HW * (HW / 4))>>>(corr, p1, (long)Bn * HW, H, Wd);
    k_avgpool2<<<ELW((long)Bn * HW * (HW / 16))>>>(p1, p2, (long)Bn * HW, H / 2, Wd / 2);
    k_avgpool2<<<ELW((long)Bn * HW * (HW / 64))>>>(p2, p3, (long)Bn * HW, H / 4, Wd / 4);

    // net / inp
    k_split_tanh_relu<<<ELW((long)Bn * 128 * HW)>>>(cn, net, inp, Bn, HW);
    k_init_coords<<<ELW((long)Bn * 2 * HW)>>>(coords, Bn, H, Wd);

    const long n128 = (long)Bn * 128 * HW;
    const long n2   = (long)Bn * 2 * HW;

    for (int it = 0; it < iters; ++it) {
        // correlation lookup + current flow
        k_corr_lookup<<<ELW((long)Bn * 324 * HW)>>>(corr, p1, p2, p3, coords, cfeat, Bn, H, Wd);
        k_flow_from_coords<<<ELW(n2)>>>(coords, flow, Bn, H, Wd);

        // motion encoder
        to_bf16(stream, cfeat, cfeatb, (long)Bn * 324 * HW);
        conv(stream, cfeatb, uW.cc1, U.cc1_b, c256, Bn, 324, H, Wd, 256, 1, 1, 1, 0, 0, 1, 1.f);
        to_bf16(stream, c256, tb256, (long)Bn * 256 * HW);
        conv(stream, tb256, uW.cc2, U.cc2_b, c192, Bn, 256, H, Wd, 192, 3, 3, 1, 1, 1, 1, 1.f);
        to_bf16(stream, flow, flowb, n2);
        conv(stream, flowb, uW.cf1, U.cf1_b, fbuf128, Bn, 2, H, Wd, 128, 7, 7, 1, 3, 3, 1, 1.f);
        to_bf16(stream, fbuf128, tb128, n128);
        conv(stream, tb128, uW.cf2, U.cf2_b, fbuf64, Bn, 128, H, Wd, 64, 3, 3, 1, 1, 1, 1, 1.f);
        k_concat2_bf16<<<ELW((long)Bn * 256 * HW)>>>(c192, fbuf64, cat256b, Bn, 192, 64, HW);
        conv(stream, cat256b, uW.cm, U.cm_b, m126, Bn, 256, H, Wd, 126, 3, 3, 1, 1, 1, 1, 1.f);

        // x = concat(inp, motion) = concat(inp, m126, flow): 256 ch
        k_concat3_f32<<<ELW((long)Bn * 256 * HW)>>>(inp, m126, flow, xf, Bn, 128, 126, 2, HW);

        // separable GRU: horizontal (1x5) then vertical (5x1)
        for (int a = 0; a < 2; ++a) {
            const __bf16 *wz = a ? uW.zv : uW.zh, *wr = a ? uW.rv : uW.rh, *wq = a ? uW.qv : uW.qh;
            const float *bz = a ? U.zv_b : U.zh_b, *br = a ? U.rv_b : U.rh_b, *bq = a ? U.qv_b : U.qh_b;
            int KHk = a ? 5 : 1, KWk = a ? 1 : 5, pH = a ? 2 : 0, pW = a ? 0 : 2;
            k_concat2_bf16<<<ELW((long)Bn * 384 * HW)>>>(net, xf, hxb, Bn, 128, 256, HW);
            conv(stream, hxb, wz, bz, zb, Bn, 384, H, Wd, 128, KHk, KWk, 1, pH, pW, 0, 1.f);
            k_sigmoid<<<ELW(n128)>>>(zb, n128);
            conv(stream, hxb, wr, br, rb, Bn, 384, H, Wd, 128, KHk, KWk, 1, pH, pW, 0, 1.f);
            k_sigmoid<<<ELW(n128)>>>(rb, n128);
            k_mul<<<ELW(n128)>>>(rb, net, rhb, n128);
            k_concat2_bf16<<<ELW((long)Bn * 384 * HW)>>>(rhb, xf, hxb, Bn, 128, 256, HW);
            conv(stream, hxb, wq, bq, qb, Bn, 384, H, Wd, 128, KHk, KWk, 1, pH, pW, 0, 1.f);
            k_tanh<<<ELW(n128)>>>(qb, n128);
            k_gru_combine<<<ELW(n128)>>>(net, zb, qb, n128);
        }

        // flow head + mask head
        to_bf16(stream, net, netb, n128);
        conv(stream, netb, uW.fh1, U.fh1_b, t256, Bn, 128, H, Wd, 256, 3, 3, 1, 1, 1, 1, 1.f);
        to_bf16(stream, t256, tb256, (long)Bn * 256 * HW);
        conv(stream, tb256, uW.fh2, U.fh2_b, delta, Bn, 256, H, Wd, 2, 3, 3, 1, 1, 1, 0, 1.f);
        conv(stream, netb, uW.mk1, U.mk1_b, t256, Bn, 128, H, Wd, 256, 3, 3, 1, 1, 1, 1, 1.f);
        to_bf16(stream, t256, tb256, (long)Bn * 256 * HW);
        conv(stream, tb256, uW.mk2, U.mk2_b, mask, Bn, 256, H, Wd, 576, 1, 1, 1, 0, 0, 0, 0.25f);

        k_add<<<ELW(n2)>>>(coords, delta, n2);
    }

    // final convex upsampling (uses last iteration's mask)
    k_flow_from_coords<<<ELW(n2)>>>(coords, flow, Bn, H, Wd);
    k_upsample<<<ELW((long)Bn * 2 * 8 * H * 8 * Wd)>>>(flow, mask, (float*)d_out, Bn, H, Wd);
}
